// NNWindowAttention_39702677684798
// MI455X (gfx1250) — matrix operand order repaired and checked by reading, not compiled, not run on hardware
//
#include <hip/hip_runtime.h>
#include <hip/hip_bf16.h>

typedef __attribute__((ext_vector_type(16))) _Float16 v16h;
typedef __attribute__((ext_vector_type(8)))  _Float16 v8h;
typedef __attribute__((ext_vector_type(8)))  float    v8f;

#define NH    12
#define HD    32
#define NTOK  98
#define DIMC  384
#define NWIN  64
#define NPADQ 112
#define NPADK 128
#define SCALE 0.17677669529663687f   // 32^-0.5

// ---------------------------------------------------------------------------
// Fragment loaders for CDNA5 WMMA (wave32); A and B share the same contiguous k order.
// A (16x32 f16, MxK): lane l: m=l&15, half=l>>4; holds K = half*16..+15 (contiguous)
// B (32x16 f16, KxN): lane l: n=l&15, half=l>>4; holds K = half*16..+15 (contiguous);
//   source is the transposed operand stored row-major (row = n, leading dim ld).
// C/D (16x16 f32):    lane l: n=l&15; reg r: m = r + 8*(l>>4)
// ---------------------------------------------------------------------------
__device__ __forceinline__ v16h load_a_frag(const _Float16* p, int ld) {
    int lane = threadIdx.x & 31;
    int m = lane & 15, half = lane >> 4;
    const _Float16* r0 = p + m * ld + half * 16;
    v16h a;
#pragma unroll
    for (int t = 0; t < 16; ++t) a[t] = r0[t];
    return a;
}

__device__ __forceinline__ v16h load_b_frag(const _Float16* p, int ld) {
    int lane = threadIdx.x & 31;
    int n = lane & 15, half = lane >> 4;
    const _Float16* r0 = p + n * ld + half * 16;
    v16h b;
#pragma unroll
    for (int t = 0; t < 16; ++t) b[t] = r0[t];
    return b;
}

__device__ __forceinline__ v8f wmma_f16(v16h a, v16h b, v8f c) {
    return __builtin_amdgcn_wmma_f32_16x16x32_f16(false, a, false, b, (short)0, c, false, false);
}

// ---------------------------------------------------------------------------
// Prep: convert qkv_w (1152x384) and proj_w (384x384) to f16 once.
// ---------------------------------------------------------------------------
__global__ void k_prep(const float* __restrict__ qkvw, const float* __restrict__ projw,
                       _Float16* __restrict__ qkvwh, _Float16* __restrict__ projwh) {
    int idx = blockIdx.x * 256 + threadIdx.x;
    if (idx < 1152 * 384) qkvwh[idx] = (_Float16)qkvw[idx];
    int idx2 = idx - 1152 * 384;
    if (idx2 >= 0 && idx2 < 384 * 384) projwh[idx2] = (_Float16)projw[idx2];
}

// ---------------------------------------------------------------------------
// Combine relative-position bias (gather) + window mask:
// comb[(w*12+h)*98*98 + i*98 + j]  (f32, 29.5 MB -> L2 resident)
// ---------------------------------------------------------------------------
__global__ void k_bias(const float* __restrict__ mask, const float* __restrict__ rpb,
                       const int* __restrict__ ridx, float* __restrict__ comb) {
    int idx = blockIdx.x * 256 + threadIdx.x;
    const int total = NWIN * NH * NTOK * NTOK;
    if (idx >= total) return;
    int j = idx % NTOK;
    int i = (idx / NTOK) % NTOK;
    int rem = idx / (NTOK * NTOK);
    int h = rem % NH;
    int w = rem / NH;
    comb[idx] = rpb[ridx[i * NTOK + j] * NH + h] + mask[(w * NTOK + i) * NTOK + j];
}

// ---------------------------------------------------------------------------
// QKV GEMM (M=100352, N=1152, K=384).
// Block = 64-row M-strip: A (64x384 f16) staged in LDS ONCE (x read exactly once
// from HBM), then loop 9 N-super-tiles of 128; 8 waves = 2(M) x 4(N), each wave
// 32x32 via 2x2 WMMA; B fragments straight from f16 weights (L2 resident).
// Epilogue scatters to head-major padded Q(scaled)/K/V^T f16 buffers.
// ---------------------------------------------------------------------------
__global__ __launch_bounds__(256) void k_qkv(const float* __restrict__ x,
                                             const _Float16* __restrict__ wh,
                                             const float* __restrict__ qkv_b,
                                             _Float16* __restrict__ qh,
                                             _Float16* __restrict__ kh,
                                             _Float16* __restrict__ vt) {
    __shared__ _Float16 As[64 * 392];   // 392-stride: 784B rows, 16B aligned, bank-skewed

    const int t = threadIdx.x;
    const int mBase = blockIdx.x * 64;
    const int wv = t >> 5, wm = wv & 1, wn = wv >> 1;
    const int lane = t & 31, nloc = lane & 15, half = lane >> 4;

    // stage A strip, f32 -> f16 (64*384 = 6144 float4, 24 per thread)
#pragma unroll 4
    for (int rep = 0; rep < 24; ++rep) {
        int id = rep * 256 + t;
        int row = id / 96, c4 = (id % 96) * 4;
        float4 v = *(const float4*)(x + (size_t)(mBase + row) * DIMC + c4);
        _Float16* dst = &As[row * 392 + c4];
        dst[0] = (_Float16)v.x; dst[1] = (_Float16)v.y;
        dst[2] = (_Float16)v.z; dst[3] = (_Float16)v.w;
    }
    __syncthreads();

    // hoist row -> (batch, token) decomposition out of the N loop
    int bArr[2][8], tokArr[2][8];
#pragma unroll
    for (int i = 0; i < 2; ++i)
#pragma unroll
        for (int r = 0; r < 8; ++r) {
            int m = mBase + wm * 32 + i * 16 + r + half * 8;
            bArr[i][r] = m / NTOK;
            tokArr[i][r] = m % NTOK;
        }

    for (int ns = 0; ns < 9; ++ns) {
        v8f acc[2][2] = {};
        for (int k0 = 0; k0 < DIMC; k0 += 32) {
            v16h a0 = load_a_frag(&As[(wm * 32) * 392 + k0], 392);
            v16h a1 = load_a_frag(&As[(wm * 32 + 16) * 392 + k0], 392);
            const _Float16* bp = wh + (size_t)(ns * 128 + wn * 32) * DIMC + k0;
            v16h b0 = load_b_frag(bp, DIMC);
            v16h b1 = load_b_frag(bp + 16 * DIMC, DIMC);
            acc[0][0] = wmma_f16(a0, b0, acc[0][0]);
            acc[0][1] = wmma_f16(a0, b1, acc[0][1]);
            acc[1][0] = wmma_f16(a1, b0, acc[1][0]);
            acc[1][1] = wmma_f16(a1, b1, acc[1][1]);
        }
        // epilogue: +bias, scatter to Q (scaled) / K / V^T
#pragma unroll
        for (int i = 0; i < 2; ++i) {
#pragma unroll
            for (int j = 0; j < 2; ++j) {
                int ncol = ns * 128 + wn * 32 + j * 16 + nloc;
                float bias = qkv_b[ncol];
                int which = ncol / DIMC;         // 0=q 1=k 2=v
                int rem = ncol % DIMC;
                int h = rem >> 5, d = rem & 31;
#pragma unroll
                for (int r = 0; r < 8; ++r) {
                    int b = bArr[i][r], tok = tokArr[i][r];
                    float v = acc[i][j][r] + bias;
                    if (which == 0)
                        qh[((size_t)(b * NH + h) * NPADQ + tok) * HD + d] = (_Float16)(v * SCALE);
                    else if (which == 1)
                        kh[((size_t)(b * NH + h) * NPADK + tok) * HD + d] = (_Float16)v;
                    else
                        vt[((size_t)(b * NH + h) * HD + d) * NPADK + tok] = (_Float16)v;
                }
            }
        }
    }
}

// ---------------------------------------------------------------------------
// Attention: one block per (window b, head h); 7 waves, one per 16-row query
// tile. K (128x32) and V^T (32x128) staged once in LDS; per wave: 8 WMMAs for
// S=QK^T, bias/mask add (-1e30 on padded keys), row softmax via xor-shuffles
// (masks 1..8 stay inside the 16-lane half that owns rows r+8*half), P -> f16
// through LDS (reshapes D-fragments into A-fragments), 8 WMMAs for O=PV.
// ---------------------------------------------------------------------------
__global__ __launch_bounds__(224) void k_attn(const _Float16* __restrict__ qh,
                                              const _Float16* __restrict__ kh,
                                              const _Float16* __restrict__ vt,
                                              const float* __restrict__ biasc,
                                              _Float16* __restrict__ ctx) {
    __shared__ _Float16 Ks[128 * 40];       // 10.0 KB
    __shared__ _Float16 Vs[32 * 136];       //  8.5 KB
    __shared__ _Float16 lp[7 * 16 * 136];   // 29.8 KB   (total < 64 KB)

    const int id = blockIdx.x;
    const int h = id % NH;
    const int b = id / NH;
    const int w = b & (NWIN - 1);
    const int t = threadIdx.x;
    const int qt = t >> 5;                       // wave = query tile
    const int lane = t & 31, nloc = lane & 15, half = lane >> 4;

    const _Float16* Kg = kh + (size_t)(b * NH + h) * NPADK * HD;
    const _Float16* Vg = vt + (size_t)(b * NH + h) * HD * NPADK;

    // cooperative stage of K and V^T (1024 chunks of 8 halves)
    for (int c = t; c < 1024; c += 224) {
        if (c < 512) {
            int row = c >> 2, c8 = (c & 3) * 8;
            *(v8h*)&Ks[row * 40 + c8] = *(const v8h*)(Kg + row * HD + c8);
        } else {
            int cc = c - 512;
            int row = cc >> 4, c8 = (cc & 15) * 8;
            *(v8h*)&Vs[row * 136 + c8] = *(const v8h*)(Vg + row * NPADK + c8);
        }
    }
    __syncthreads();

    const _Float16* Q = qh + ((size_t)(b * NH + h) * NPADQ + qt * 16) * HD;
    const float* BS = biasc + (size_t)(w * NH + h) * NTOK * NTOK;
    _Float16* myP = &lp[qt * 16 * 136];

    v16h aq = load_a_frag(Q, HD);

    v8f S[8];
#pragma unroll
    for (int jt = 0; jt < 8; ++jt) {
        v16h bk = load_b_frag(&Ks[jt * 16 * 40], 40);
        v8f z = {};
        S[jt] = wmma_f16(aq, bk, z);
    }

    // + (rpb + mask); padded keys -> -inf
#pragma unroll
    for (int jt = 0; jt < 8; ++jt) {
        int j = jt * 16 + nloc;
#pragma unroll
        for (int r = 0; r < 8; ++r) {
            int i = qt * 16 + r + half * 8;
            int ib = i < NTOK ? i : NTOK - 1;
            S[jt][r] = (j < NTOK) ? (S[jt][r] + BS[ib * NTOK + j]) : -1.0e30f;
        }
    }

    // row softmax
#pragma unroll
    for (int r = 0; r < 8; ++r) {
        float mx = -3.0e38f;
#pragma unroll
        for (int jt = 0; jt < 8; ++jt) mx = fmaxf(mx, S[jt][r]);
#pragma unroll
        for (int off = 1; off < 16; off <<= 1) mx = fmaxf(mx, __shfl_xor(mx, off, 32));
        float sum = 0.f;
#pragma unroll
        for (int jt = 0; jt < 8; ++jt) { float e = __expf(S[jt][r] - mx); S[jt][r] = e; sum += e; }
#pragma unroll
        for (int off = 1; off < 16; off <<= 1) sum += __shfl_xor(sum, off, 32);
        float inv = 1.0f / sum;
#pragma unroll
        for (int jt = 0; jt < 8; ++jt)
            myP[(r + half * 8) * 136 + jt * 16 + nloc] = (_Float16)(S[jt][r] * inv);
    }
    __syncthreads();

    // O = P * V
#pragma unroll
    for (int nt = 0; nt < 2; ++nt) {
        v8f acc = {};
#pragma unroll
        for (int c = 0; c < 4; ++c) {
            v16h ap = load_a_frag(&myP[c * 32], 136);
            v16h bv = load_b_frag(&Vs[nt * 16 * 136 + c * 32], 136);
            acc = wmma_f16(ap, bv, acc);
        }
#pragma unroll
        for (int r = 0; r < 8; ++r) {
            int i = qt * 16 + r + half * 8;
            if (i < NTOK)
                ctx[((size_t)b * NTOK + i) * DIMC + h * HD + nt * 16 + nloc] = (_Float16)acc[r];
        }
    }
}

// ---------------------------------------------------------------------------
// Projection GEMM (M=100352, N=K=384): 64-row A-strip (already f16) staged
// once in LDS, 3 N-super-tiles of 128; f32 output + bias.
// ---------------------------------------------------------------------------
__global__ __launch_bounds__(256) void k_proj(const _Float16* __restrict__ ctx,
                                              const _Float16* __restrict__ wh,
                                              const float* __restrict__ pb,
                                              float* __restrict__ out) {
    __shared__ _Float16 As[64 * 392];

    const int t = threadIdx.x;
    const int mBase = blockIdx.x * 64;
    const int wv = t >> 5, wm = wv & 1, wn = wv >> 1;
    const int lane = t & 31, nloc = lane & 15, half = lane >> 4;

    // stage A strip (64*384 halves = 3072 v8h chunks, 12 per thread)
#pragma unroll 4
    for (int rep = 0; rep < 12; ++rep) {
        int id = rep * 256 + t;
        int row = id / 48, c8 = (id % 48) * 8;
        v8h v = *(const v8h*)(ctx + (size_t)(mBase + row) * DIMC + c8);
        *(v8h*)&As[row * 392 + c8] = v;
    }
    __syncthreads();

    for (int ns = 0; ns < 3; ++ns) {
        v8f acc[2][2] = {};
        for (int k0 = 0; k0 < DIMC; k0 += 32) {
            v16h a0 = load_a_frag(&As[(wm * 32) * 392 + k0], 392);
            v16h a1 = load_a_frag(&As[(wm * 32 + 16) * 392 + k0], 392);
            const _Float16* bp = wh + (size_t)(ns * 128 + wn * 32) * DIMC + k0;
            v16h b0 = load_b_frag(bp, DIMC);
            v16h b1 = load_b_frag(bp + 16 * DIMC, DIMC);
            acc[0][0] = wmma_f16(a0, b0, acc[0][0]);
            acc[0][1] = wmma_f16(a0, b1, acc[0][1]);
            acc[1][0] = wmma_f16(a1, b0, acc[1][0]);
            acc[1][1] = wmma_f16(a1, b1, acc[1][1]);
        }
#pragma unroll
        for (int i = 0; i < 2; ++i) {
#pragma unroll
            for (int j = 0; j < 2; ++j) {
                int ncol = ns * 128 + wn * 32 + j * 16 + nloc;
                float bias = pb[ncol];
#pragma unroll
                for (int r = 0; r < 8; ++r) {
                    int m = mBase + wm * 32 + i * 16 + r + half * 8;
                    out[(size_t)m * DIMC + ncol] = acc[i][j][r] + bias;
                }
            }
        }
    }
}

// ---------------------------------------------------------------------------
// Workspace layout (bytes, all 256-aligned):
//   Qh   f16 (1024,12,112,32)  @ 0          88,080,384
//   Kh   f16 (1024,12,128,32)  @ 88080384   100,663,296
//   Vt   f16 (1024,12,32,128)  @ 188743680  100,663,296
//   comb f32 (64,12,98,98)     @ 289406976  29,503,488
//   ctx  f16 (100352,384)      @ 318910464  77,070,336
//   qkvw f16 (1152,384)        @ 395980800  884,736
//   projw f16 (384,384)        @ 396865536  294,912    (end 397,160,448)
// ---------------------------------------------------------------------------
extern "C" void kernel_launch(void* const* d_in, const int* in_sizes, int n_in,
                              void* d_out, int out_size, void* d_ws, size_t ws_size,
                              hipStream_t stream) {
    const float* x     = (const float*)d_in[0];
    const float* mask  = (const float*)d_in[1];
    const float* rpb   = (const float*)d_in[2];
    const float* qkvw  = (const float*)d_in[3];
    const float* qkvb  = (const float*)d_in[4];
    const float* projw = (const float*)d_in[5];
    const float* projb = (const float*)d_in[6];
    const int*   ridx  = (const int*)d_in[7];

    char* ws = (char*)d_ws;
    _Float16* qh     = (_Float16*)(ws);
    _Float16* kh     = (_Float16*)(ws + 88080384);
    _Float16* vt     = (_Float16*)(ws + 188743680);
    float*    comb   = (float*)   (ws + 289406976);
    _Float16* ctx    = (_Float16*)(ws + 318910464);
    _Float16* qkvwh  = (_Float16*)(ws + 395980800);
    _Float16* projwh = (_Float16*)(ws + 396865536);
    float* out = (float*)d_out;

    k_prep<<<2304, 256, 0, stream>>>(qkvw, projw, qkvwh, projwh);
    k_bias<<<28812, 256, 0, stream>>>(mask, rpb, ridx, comb);
    k_qkv<<<1568, 256, 0, stream>>>(x, qkvwh, qkvb, qh, kh, vt);
    k_attn<<<1024 * NH, 224, 0, stream>>>(qh, kh, vt, comb, ctx);
    k_proj<<<1568, 256, 0, stream>>>(ctx, projwh, projb, out);
}
